// NeuralCDE_50869592655458
// MI455X (gfx1250) — matrix operand order repaired and checked by reading, not compiled, not run on hardware
//
#include <hip/hip_runtime.h>
#include <hip/hip_bf16.h>

// Neural CDE forward, persistent-weight WMMA formulation for gfx1250.
//
// B=512 samples -> 32 workgroups of 16 samples (WMMA N=16), 256 threads
// (8 wave32). Vector-field MLP weights are bf16 register-resident WMMA
// A-fragments for the entire T=1024 step loop; biases are folded into the
// WMMA accumulator initializer (C layout row == bias index). Activations
// move through LDS transposed ([sample][feature]) so B fragments are
// two contiguous 16B/lane ds loads and C writeback is packed stores. All
// activation functions are branchless using raw v_exp/v_log/v_rcp.

#define B_  512
#define T_  1024
#define D_  8
#define H_  64
#define W_  128
#define NB  16
#define NTHREADS 256

typedef __bf16 bf16_t;
typedef __attribute__((ext_vector_type(16))) __bf16 v16bf;
typedef __attribute__((ext_vector_type(8)))  __bf16 v8bf;
typedef __attribute__((ext_vector_type(8)))  float  v8f;

// ---- raw hardware transcendentals (branchless, no denorm fix-up) ----
__device__ __forceinline__ float exp2_fast(float x) {
#if __has_builtin(__builtin_amdgcn_exp2f)
  return __builtin_amdgcn_exp2f(x);
#else
  return __expf(x * 0.6931471805599453f);
#endif
}
__device__ __forceinline__ float log2_fast(float x) {   // arg >= 1 here
#if __has_builtin(__builtin_amdgcn_logf)
  return __builtin_amdgcn_logf(x);
#else
  return __logf(x) * 1.4426950408889634f;
#endif
}
__device__ __forceinline__ float rcp_fast(float x) {
#if __has_builtin(__builtin_amdgcn_rcpf)
  return __builtin_amdgcn_rcpf(x);
#else
  return 1.f / x;
#endif
}

// softplus(x) = max(x,0) + ln(1 + 2^(-|x|*log2e)); log arg in [1,2] -> safe.
__device__ __forceinline__ float softplusf(float x) {
  const float u = exp2_fast(-fabsf(x) * 1.442695041f);
  return fmaxf(x, 0.f) + 0.693147181f * log2_fast(1.f + u);
}

// tanh: hardware builtin if present, else exp2/rcp form (inf -> r = 1 exact).
__device__ __forceinline__ float tanh_fast(float x) {
#if __has_builtin(__builtin_amdgcn_tanhf)
  return __builtin_amdgcn_tanhf(x);
#elif __has_builtin(__builtin_amdgcn_tanh_f32)
  return __builtin_amdgcn_tanh_f32(x);
#else
  const float e = exp2_fast(fabsf(x) * 2.885390082f);   // exp(2|x|)
  const float r = 1.f - 2.f * rcp_fast(e + 1.f);
  return copysignf(r, x);
#endif
}

// A fragment (16x32 bf16, M x K) for V_WMMA_F32_16X16X32_BF16.
// Lane L<16 : row M=mbase+L,      holds K = kb+{0..7} and kb+{16..23}
// Lane L>=16: row M=mbase+(L-16), holds K = kb+{8..15} and kb+{24..31}
__device__ __forceinline__ v16bf load_a_frag(const float* Wt, int ncols,
                                             int mbase, int kb, int lane) {
  const int row = mbase + (lane & 15);
  const int k0  = kb + ((lane >> 4) & 1) * 8;
  const float* p = Wt + (size_t)row * ncols + k0;
  const float4 f0 = *(const float4*)(p + 0);
  const float4 f1 = *(const float4*)(p + 4);
  const float4 f2 = *(const float4*)(p + 16);
  const float4 f3 = *(const float4*)(p + 20);
  v16bf a;
  a[0]=(__bf16)f0.x;  a[1]=(__bf16)f0.y;  a[2]=(__bf16)f0.z;  a[3]=(__bf16)f0.w;
  a[4]=(__bf16)f1.x;  a[5]=(__bf16)f1.y;  a[6]=(__bf16)f1.z;  a[7]=(__bf16)f1.w;
  a[8]=(__bf16)f2.x;  a[9]=(__bf16)f2.y;  a[10]=(__bf16)f2.z; a[11]=(__bf16)f2.w;
  a[12]=(__bf16)f3.x; a[13]=(__bf16)f3.y; a[14]=(__bf16)f3.z; a[15]=(__bf16)f3.w;
  return a;
}

// Per-lane bias slice from LDS; serves directly as the WMMA C initializer
// (C layout: VGPR r of lanes 0-15 -> M=mb+r; lanes 16-31 -> M=mb+8+r).
// Lanes 0-15 read one 32B chunk, lanes 16-31 another -> LDS broadcasts.
__device__ __forceinline__ v8f load_bias_lds(const float* bS, int mtile_base, int lane) {
  const int mb = mtile_base + (((lane >> 4) & 1) ? 8 : 0);
  const float4 x = *(const float4*)&bS[mb];
  const float4 y = *(const float4*)&bS[mb + 4];
  v8f r;
  r[0]=x.x; r[1]=x.y; r[2]=x.z; r[3]=x.w;
  r[4]=y.x; r[5]=y.y; r[6]=y.z; r[7]=y.w;
  return r;
}

// B fragment (32x16 bf16, K x N) from transposed LDS activations hT[n][k].
// Lane L<16 : col N=L,    holds K = kb+{0..7} and kb+{16..23}
// Lane L>=16: col N=L-16, holds K = kb+{8..15} and kb+{24..31}
template <int STRIDE>
__device__ __forceinline__ v16bf load_b_frag(bf16_t (*hT)[STRIDE], int kb, int lane) {
  const int n  = lane & 15;
  const int ko = kb + ((lane >> 4) & 1) * 8;
  const v8bf b0 = *(const v8bf*)&hT[n][ko];
  const v8bf b1 = *(const v8bf*)&hT[n][ko + 16];
  v16bf b;
#pragma unroll
  for (int i = 0; i < 8; ++i) { b[i] = b0[i]; b[i + 8] = b1[i]; }
  return b;
}

// Softplus + pack to bf16, one 16B ds store per lane.
template <int STRIDE>
__device__ __forceinline__ void store_c_softplus(bf16_t (*hT)[STRIDE], int mtile_base,
                                                 const v8f& c, int lane) {
  const int n  = lane & 15;
  const int mb = mtile_base + (((lane >> 4) & 1) ? 8 : 0);
  v8bf pk;
#pragma unroll
  for (int r = 0; r < 8; ++r) pk[r] = (__bf16)softplusf(c[r]);
  *(v8bf*)&hT[n][mb] = pk;
}

// Tanh, kept in f32 (consumed by the VALU Euler update, not by WMMA).
template <int STRIDE>
__device__ __forceinline__ void store_c_tanh(float (*hT)[STRIDE], int mtile_base,
                                             const v8f& c, int lane) {
  const int n  = lane & 15;
  const int mb = mtile_base + (((lane >> 4) & 1) ? 8 : 0);
  float4 lo, hi;
  lo.x = tanh_fast(c[0]); lo.y = tanh_fast(c[1]);
  lo.z = tanh_fast(c[2]); lo.w = tanh_fast(c[3]);
  hi.x = tanh_fast(c[4]); hi.y = tanh_fast(c[5]);
  hi.z = tanh_fast(c[6]); hi.w = tanh_fast(c[7]);
  *(float4*)&hT[n][mb]     = lo;
  *(float4*)&hT[n][mb + 4] = hi;
}

__global__ __launch_bounds__(NTHREADS, 1)
void neural_cde_kernel(const float* __restrict__ ts,
                       const float* __restrict__ cd, const float* __restrict__ cc,
                       const float* __restrict__ cb, const float* __restrict__ ca,
                       const float* __restrict__ Wi0, const float* __restrict__ bi0,
                       const float* __restrict__ Wi1, const float* __restrict__ bi1,
                       const float* __restrict__ Wi2, const float* __restrict__ bi2,
                       const float* __restrict__ Wf0, const float* __restrict__ bf0,
                       const float* __restrict__ Wf1, const float* __restrict__ bf1,
                       const float* __restrict__ Wf2, const float* __restrict__ bf2,
                       const float* __restrict__ Wl,  const float* __restrict__ bl,
                       float* __restrict__ out) {
  // Transposed activations, rows padded to 16B multiples.
  __shared__ __align__(32) bf16_t yT [NB][H_ + 8];          // bf16: WMMA B input
  __shared__ __align__(32) bf16_t h0T[NB][W_ + 8];
  __shared__ __align__(32) bf16_t h1T[NB][W_ + 8];
  __shared__ __align__(32) float  vfTf[NB][H_ * D_ + 4];    // f32: VALU consumer
  __shared__ float yF[NB][H_];
  __shared__ __align__(16) float dxS[NB][D_];
  __shared__ float dtv[NB];
  __shared__ __align__(16) float bf0S[W_], bf1S[W_], bf2S[H_ * D_];
  __shared__ float WlS[H_];
  __shared__ float hA[NB][W_], hB[NB][W_];
  __shared__ float x0S[NB][D_];

  const int tid  = threadIdx.x;
  const int lane = tid & 31;
  const int wave = tid >> 5;
  const int s0   = blockIdx.x * NB;

  // ---- stage biases / readout weights / per-sample dt ----
  for (int i = tid; i < W_; i += NTHREADS) { bf0S[i] = bf0[i]; bf1S[i] = bf1[i]; }
  for (int i = tid; i < H_ * D_; i += NTHREADS) bf2S[i] = bf2[i];
  for (int i = tid; i < H_; i += NTHREADS) WlS[i] = Wl[i];
  if (tid < NB) {
    const size_t s = (size_t)(s0 + tid);
    dtv[tid] = ts[s * T_ + 1] - ts[s * T_];
  }
  if (tid < NB * D_) {
    const int n = tid / D_, d = tid % D_;
    x0S[n][d] = ca[(size_t)(s0 + n) * (T_ - 1) * D_ + d];   // control at t0
  }
  __syncthreads();

  // ---- one-time init MLP (relu, relu, identity): y0 = init_mlp(ca[:,0]) ----
  for (int task = tid; task < NB * W_; task += NTHREADS) {
    const int n = task / W_, o = task % W_;
    float acc = bi0[o];
#pragma unroll
    for (int d = 0; d < D_; ++d) acc += Wi0[o * D_ + d] * x0S[n][d];
    hA[n][o] = fmaxf(acc, 0.f);
  }
  __syncthreads();
  for (int task = tid; task < NB * W_; task += NTHREADS) {
    const int n = task / W_, o = task % W_;
    float acc = bi1[o];
    for (int j = 0; j < W_; ++j) acc += Wi1[o * W_ + j] * hA[n][j];
    hB[n][o] = fmaxf(acc, 0.f);
  }
  __syncthreads();
  for (int task = tid; task < NB * H_; task += NTHREADS) {
    const int n = task / H_, o = task % H_;
    float acc = bi2[o];
    for (int j = 0; j < W_; ++j) acc += Wi2[o * W_ + j] * hB[n][j];
    yF[n][o] = acc;
    yT[n][o] = (__bf16)acc;
  }

  // ---- persistent register-resident bf16 A-fragments ----
  v16bf A0[2], A1[4], A2[4][4];
#pragma unroll
  for (int kf = 0; kf < 2; ++kf) A0[kf] = load_a_frag(Wf0, H_, wave * 16, kf * 32, lane);
#pragma unroll
  for (int kf = 0; kf < 4; ++kf) A1[kf] = load_a_frag(Wf1, W_, wave * 16, kf * 32, lane);
#pragma unroll
  for (int mt = 0; mt < 4; ++mt)
#pragma unroll
    for (int kf = 0; kf < 4; ++kf)
      A2[mt][kf] = load_a_frag(Wf2, W_, (wave * 4 + mt) * 16, kf * 32, lane);

  const float blv = bl[0];

  // Per-thread time state for the merged readout/dX stage (threads 0..127:
  // 8 lanes per sample; lane-group d = tid&7 is also the dX channel).
  const int    myn = tid >> 3;
  const int    myd = tid & 7;
  const size_t myS = (size_t)(s0 + myn);
  float t0r = 0.f, dtr = 0.f;
  if (tid < NB * D_) {
    t0r = ts[myS * T_];
    dtr = ts[myS * T_ + 1] - ts[myS * T_];
    // dX for step 0 (i = 0, frac = t0 - ts[0])
    const float frac = t0r - ts[myS * T_];
    const size_t cidx = myS * (T_ - 1) * D_ + myd;
    dxS[myn][myd] = cb[cidx] + frac * (2.f * cc[cidx] + 3.f * frac * cd[cidx]);
  }
  __syncthreads();

  // ======================= T sequential Euler steps =======================
  for (int k = 0; k < T_; ++k) {
    // L0: h0 = softplus(Wf0 @ y + bf0)    [128x64 @ 64x16]
    {
      v8f c = load_bias_lds(bf0S, wave * 16, lane);   // bias as C initializer
#pragma unroll
      for (int kf = 0; kf < 2; ++kf) {
        v16bf b = load_b_frag<H_ + 8>(yT, kf * 32, lane);
        c = __builtin_amdgcn_wmma_f32_16x16x32_bf16(false, A0[kf], false, b,
                                                    (short)0, c, false, false);
      }
      store_c_softplus<W_ + 8>(h0T, wave * 16, c, lane);
    }
    __syncthreads();

    // L1: h1 = softplus(Wf1 @ h0 + bf1)   [128x128 @ 128x16]
    {
      v8f c = load_bias_lds(bf1S, wave * 16, lane);
#pragma unroll
      for (int kf = 0; kf < 4; ++kf) {
        v16bf b = load_b_frag<W_ + 8>(h0T, kf * 32, lane);
        c = __builtin_amdgcn_wmma_f32_16x16x32_bf16(false, A1[kf], false, b,
                                                    (short)0, c, false, false);
      }
      store_c_softplus<W_ + 8>(h1T, wave * 16, c, lane);
    }
    __syncthreads();

    // L2: vf = tanh(Wf2 @ h1 + bf2)       [512x128 @ 128x16]
#pragma unroll
    for (int mt = 0; mt < 4; ++mt) {
      v8f c = load_bias_lds(bf2S, (wave * 4 + mt) * 16, lane);
#pragma unroll
      for (int kf = 0; kf < 4; ++kf) {
        v16bf b = load_b_frag<W_ + 8>(h1T, kf * 32, lane);
        c = __builtin_amdgcn_wmma_f32_16x16x32_bf16(false, A2[mt][kf], false, b,
                                                    (short)0, c, false, false);
      }
      store_c_tanh<H_ * D_ + 4>(vfTf, (wave * 4 + mt) * 16, c, lane);
    }
    __syncthreads();

    // Euler update: y += dt * (vf.reshape(H,D) @ dX), fp32 master state
#pragma unroll
    for (int q = 0; q < 4; ++q) {
      const int task = tid + q * NTHREADS;
      const int n = task >> 6, h = task & 63;
      const float4 v0  = *(const float4*)&vfTf[n][h * D_];
      const float4 v1  = *(const float4*)&vfTf[n][h * D_ + 4];
      const float4 dx0 = *(const float4*)&dxS[n][0];
      const float4 dx1 = *(const float4*)&dxS[n][4];
      float acc = v0.x * dx0.x + v0.y * dx0.y + v0.z * dx0.z + v0.w * dx0.w +
                  v1.x * dx1.x + v1.y * dx1.y + v1.z * dx1.z + v1.w * dx1.w;
      const float yn = yF[n][h] + dtv[n] * acc;
      yF[n][h] = yn;
      yT[n][h] = (__bf16)yn;
    }
    __syncthreads();

    // Merged stage: per-step sigmoid readout (shuffle-reduced over 8 lanes
    // per sample) + time advance + dX for the next step + prefetch.
    if (tid < NB * D_) {
      const int hb = myd * 8;
      float acc = 0.f;
#pragma unroll
      for (int j = 0; j < 8; ++j) acc += yF[myn][hb + j] * WlS[hb + j];
      acc += __shfl_xor(acc, 1);
      acc += __shfl_xor(acc, 2);
      acc += __shfl_xor(acc, 4);
      if (myd == 0)
        out[myS * T_ + k] = rcp_fast(1.f + exp2_fast(-(acc + blv) * 1.442695041f));

      t0r += dtr;
      if (k + 1 < T_) {
        const int i2 = (k + 1 < T_ - 1) ? (k + 1) : (T_ - 2);
        const float frac = t0r - ts[myS * T_ + i2];
        const size_t cidx = (myS * (T_ - 1) + i2) * D_ + myd;
        dxS[myn][myd] = cb[cidx] + frac * (2.f * cc[cidx] + 3.f * frac * cd[cidx]);
        if (i2 < T_ - 2) {
          __builtin_prefetch(&cb[cidx + D_], 0, 3);
          __builtin_prefetch(&cc[cidx + D_], 0, 3);
          __builtin_prefetch(&cd[cidx + D_], 0, 3);
        }
      }
    }
    __syncthreads();
  }
}

extern "C" void kernel_launch(void* const* d_in, const int* in_sizes, int n_in,
                              void* d_out, int out_size, void* d_ws, size_t ws_size,
                              hipStream_t stream) {
  const float* ts  = (const float*)d_in[0];
  const float* cd  = (const float*)d_in[1];
  const float* cc  = (const float*)d_in[2];
  const float* cb  = (const float*)d_in[3];
  const float* ca  = (const float*)d_in[4];
  const float* Wi0 = (const float*)d_in[5];
  const float* bi0 = (const float*)d_in[6];
  const float* Wi1 = (const float*)d_in[7];
  const float* bi1 = (const float*)d_in[8];
  const float* Wi2 = (const float*)d_in[9];
  const float* bi2 = (const float*)d_in[10];
  const float* Wf0 = (const float*)d_in[11];
  const float* bf0 = (const float*)d_in[12];
  const float* Wf1 = (const float*)d_in[13];
  const float* bf1 = (const float*)d_in[14];
  const float* Wf2 = (const float*)d_in[15];
  const float* bf2 = (const float*)d_in[16];
  const float* Wl  = (const float*)d_in[17];
  const float* bl  = (const float*)d_in[18];
  float* out = (float*)d_out;

  dim3 grid(B_ / NB);      // 32 workgroups x 16 samples
  dim3 block(NTHREADS);    // 8 wave32
  hipLaunchKernelGGL(neural_cde_kernel, grid, block, 0, stream,
                     ts, cd, cc, cb, ca, Wi0, bi0, Wi1, bi1, Wi2, bi2,
                     Wf0, bf0, Wf1, bf1, Wf2, bf2, Wl, bl, out);
}
